// GroupQueryAttention_87282325390117
// MI455X (gfx1250) — matrix operand order repaired and checked by reading, not compiled, not run on hardware
//
#include <hip/hip_runtime.h>
#include <hip/hip_bf16.h>

// ---------------------------------------------------------------------------
// GQA forward for MI455X (gfx1250, wave32).
//  * all matmuls: v_wmma_f32_16x16x32_bf16 (bf16 in, f32 accumulate)
//  * global->LDS tile movement: global_load_async_to_lds_b128 (ASYNCcnt),
//    double-buffered so DMA overlaps WMMA
//  * bf16 conversion hoisted out of all hot loops
// ---------------------------------------------------------------------------

typedef __attribute__((ext_vector_type(16))) __bf16        v16bf;
typedef __attribute__((ext_vector_type(8)))  float         v8f;
typedef __attribute__((ext_vector_type(4)))  unsigned int  u32x4;
typedef __attribute__((ext_vector_type(2)))  unsigned int  u32x2;
typedef __attribute__((ext_vector_type(4)))  float         f32x4;

union Frag16 { v16bf v; unsigned short u[16]; u32x4 q[2]; };

__device__ __forceinline__ unsigned short f2bf(float x) {
    unsigned int u = __builtin_bit_cast(unsigned int, x);
    u += 0x7FFFu + ((u >> 16) & 1u);          // round-to-nearest-even
    return (unsigned short)(u >> 16);
}

// generic pointer to a __shared__ object: low 32 bits are the LDS byte offset
__device__ __forceinline__ unsigned lds_addr(const void* p) {
    return (unsigned)(unsigned long long)p;
}

// async DMA: copy 16 bytes per lane from global to LDS (ASYNCcnt-tracked)
__device__ __forceinline__ void async_ld_b128(unsigned lds, const void* g) {
    asm volatile("global_load_async_to_lds_b128 %0, %1, off"
                 :: "v"(lds), "v"(g) : "memory");
}
__device__ __forceinline__ void wait_async0() {
    asm volatile("s_wait_asynccnt 0" ::: "memory");
}

static constexpr int Bsz = 2, T = 2048, E = 4096, HQ = 32, HKV = 8, D = 128, G = 4;

// ---------------------------------------------------------------------------
// f32 -> bf16 bulk convert (4 elements/thread)
// ---------------------------------------------------------------------------
__global__ void cvt_f32_bf16_kernel(const float* __restrict__ src,
                                    unsigned short* __restrict__ dst, int n4)
{
    int i = blockIdx.x * blockDim.x + threadIdx.x;
    if (i >= n4) return;
    f32x4 x = *(const f32x4*)&src[(size_t)i * 4];
    u32x2 p;
    p.x = (unsigned)f2bf(x.x) | ((unsigned)f2bf(x.y) << 16);
    p.y = (unsigned)f2bf(x.z) | ((unsigned)f2bf(x.w) << 16);
    *(u32x2*)&dst[(size_t)i * 4] = p;
}

// ---------------------------------------------------------------------------
// RoPE (adjacent-pair rotation) fused with f32 -> bf16 down-convert.
// ---------------------------------------------------------------------------
__global__ void rope_to_bf16_kernel(const float* __restrict__ X,
                                    unsigned short* __restrict__ Xb,
                                    const float* __restrict__ fr,
                                    int heads, int total)
{
    int i = blockIdx.x * blockDim.x + threadIdx.x;
    if (i >= total) return;
    int pairs = heads * (D / 2);
    int row = i / pairs;                 // b*T + t
    int rem = i - row * pairs;
    int h   = rem / (D / 2);
    int p   = rem - h * (D / 2);
    int t   = row % T;
    float c = fr[t * D + 2 * p];
    float s = fr[t * D + 2 * p + 1];
    size_t base = (size_t)row * pairs * 2 + (size_t)h * D + 2 * p;
    float x0 = X[base], x1 = X[base + 1];
    float y0 = x0 * c - x1 * s;
    float y1 = x1 * c + x0 * s;
    *(unsigned*)&Xb[base] = (unsigned)f2bf(y0) | ((unsigned)f2bf(y1) << 16);
}

// ---------------------------------------------------------------------------
// NT GEMM: C[M][N] = A[M][K] * W[N][K]^T + bias[N]
// A, W bf16 row-major; C f32.  128x128 block, BK=32, 8 waves (2x4), each wave
// 64x32 = 4x2 WMMA accumulators.  Tiles DMA'd via async loads, double-buffered.
// ---------------------------------------------------------------------------
__global__ __launch_bounds__(256) void gemm_nt_bf16(
    const unsigned short* __restrict__ A, const unsigned short* __restrict__ W,
    const float* __restrict__ bias, float* __restrict__ C,
    int M, int N, int K)
{
    __shared__ unsigned short As[2][128 * 32];   // [m][k], 8 KB per buffer
    __shared__ unsigned short Ws[2][128 * 32];   // [n][k]

    const int tid  = threadIdx.x;
    const int lane = tid & 31, wid = tid >> 5;
    const int hi   = lane >> 4, lm = lane & 15;
    const int wm   = wid >> 2;                // 0..1: 64-row slice
    const int wn   = wid & 3;                 // 0..3: 32-col slice
    const int m0   = blockIdx.y * 128;
    const int n0   = blockIdx.x * 128;

    // 512 16B chunks per 128x32 tile; this thread owns chunks tid and tid+256.
    const int r0c = tid >> 2,          c0c = (tid & 3) * 8;
    const int r1c = (tid + 256) >> 2,  c1c = ((tid + 256) & 3) * 8;
    const unsigned short* ga0 = A + (size_t)(m0 + r0c) * K + c0c;
    const unsigned short* ga1 = A + (size_t)(m0 + r1c) * K + c1c;
    const unsigned short* gw0 = W + (size_t)(n0 + r0c) * K + c0c;
    const unsigned short* gw1 = W + (size_t)(n0 + r1c) * K + c1c;
    unsigned la0[2], la1[2], lw0[2], lw1[2];
    for (int bb = 0; bb < 2; ++bb) {
        la0[bb] = lds_addr(&As[bb][r0c * 32 + c0c]);
        la1[bb] = lds_addr(&As[bb][r1c * 32 + c1c]);
        lw0[bb] = lds_addr(&Ws[bb][r0c * 32 + c0c]);
        lw1[bb] = lds_addr(&Ws[bb][r1c * 32 + c1c]);
    }

    v8f acc[4][2];
    for (int i = 0; i < 4; ++i)
        for (int j = 0; j < 2; ++j)
            acc[i][j] = (v8f){0.f,0.f,0.f,0.f,0.f,0.f,0.f,0.f};

    auto issue = [&](int bb) {
        async_ld_b128(la0[bb], ga0);  async_ld_b128(la1[bb], ga1);
        async_ld_b128(lw0[bb], gw0);  async_ld_b128(lw1[bb], gw1);
        ga0 += 32; ga1 += 32; gw0 += 32; gw1 += 32;
    };

    issue(0);
    wait_async0();
    __syncthreads();

    const int nk = K / 32;
    for (int kt = 0; kt < nk; ++kt) {
        const int cur = kt & 1;
        if (kt + 1 < nk) issue(cur ^ 1);       // prefetch next tile while computing

        const unsigned short* asb = &As[cur][0];
        const unsigned short* wsb = &Ws[cur][0];

        Frag16 b[2];
        for (int jn = 0; jn < 2; ++jn) {
            const unsigned short* p = wsb + (wn * 32 + jn * 16 + lm) * 32;
            b[jn].q[0] = *(const u32x4*)(p + 8 * hi);
            b[jn].q[1] = *(const u32x4*)(p + 16 + 8 * hi);
        }
        for (int im = 0; im < 4; ++im) {
            Frag16 a;
            const unsigned short* p = asb + (wm * 64 + im * 16 + lm) * 32;
            a.q[0] = *(const u32x4*)(p + 8 * hi);
            a.q[1] = *(const u32x4*)(p + 16 + 8 * hi);
            for (int jn = 0; jn < 2; ++jn)
                acc[im][jn] = __builtin_amdgcn_wmma_f32_16x16x32_bf16(
                    false, a.v, false, b[jn].v, (short)0, acc[im][jn], false, false);
        }
        wait_async0();        // own prefetch landed
        __syncthreads();      // everyone's prefetch landed + reads of cur done
    }

    for (int im = 0; im < 4; ++im) {
        for (int jn = 0; jn < 2; ++jn) {
            int c  = n0 + wn * 32 + jn * 16 + lm;
            int r0 = m0 + wm * 64 + im * 16 + 8 * hi;
            float bv = bias[c];
            for (int j = 0; j < 8; ++j)
                C[(size_t)(r0 + j) * N + c] = acc[im][jn][j] + bv;
        }
    }
}

// ---------------------------------------------------------------------------
// Flash attention on bf16 Q/K/V: 128 queries per WG (16 per wave), stream
// 64-key tiles.  K tile via async DMA; V tile transposed manually.
// Writes bf16 context for the O-projection GEMM.
// ---------------------------------------------------------------------------
__global__ __launch_bounds__(256) void flash_attn_kernel(
    const unsigned short* __restrict__ Qb, const unsigned short* __restrict__ Kb,
    const unsigned short* __restrict__ Vb, unsigned short* __restrict__ Y)
{
    __shared__ unsigned short Ks[64 * 128];   // [key][d]   16 KB
    __shared__ unsigned short Vs[128 * 64];   // [d][key]   16 KB (transposed)
    __shared__ unsigned short Ps[8][16 * 64]; // per-wave P 16 KB

    const int tid  = threadIdx.x;
    const int lane = tid & 31, wid = tid >> 5;
    const int hi   = lane >> 4, lm = lane & 15;
    const int t0   = blockIdx.x * 128;
    const int bh   = blockIdx.y;
    const int b = bh / HQ, h = bh % HQ, kvh = h / G;

    // Persistent Q A-fragments for this wave's 16 rows (4 k-steps of 32 over D)
    Frag16 qf[4];
    {
        const unsigned short* qrow =
            Qb + ((size_t)(b * T + t0 + wid * 16 + lm)) * E + (size_t)h * D;
        for (int kk = 0; kk < 4; ++kk) {
            const unsigned short* p = qrow + kk * 32;
            for (int e = 0; e < 8; ++e) {
                qf[kk].u[e]     = p[8 * hi + e];
                qf[kk].u[8 + e] = p[16 + 8 * hi + e];
            }
        }
    }

    // async-chunk bookkeeping for K tile (1024 16B chunks, 4 per thread)
    const unsigned short* kg[4];
    unsigned kl[4];
    for (int j = 0; j < 4; ++j) {
        int i = tid + 256 * j;
        int r = i >> 4, c = (i & 15) * 8;
        kg[j] = Kb + ((size_t)(b * T + r)) * (HKV * D) + (size_t)kvh * D + c;
        kl[j] = lds_addr(&Ks[r * 128 + c]);
    }
    const unsigned short* Vg = Vb + ((size_t)(b * T)) * (HKV * D) + (size_t)kvh * D;

    float mrow[8], lrow[8];
    for (int j = 0; j < 8; ++j) { mrow[j] = -1e30f; lrow[j] = 0.f; }
    v8f Oacc[8];
    for (int n = 0; n < 8; ++n)
        Oacc[n] = (v8f){0.f,0.f,0.f,0.f,0.f,0.f,0.f,0.f};

    const float sm = 0.088388347648318447f;   // 1/sqrt(128)
    const int tmax = t0 + 127;

    for (int s0 = 0; s0 <= tmax; s0 += 64) {
        __syncthreads();                       // LDS buffers free to overwrite
        for (int j = 0; j < 4; ++j) {
            async_ld_b128(kl[j], kg[j]);       // DMA K tile
            kg[j] += (size_t)64 * (HKV * D);
        }
        for (int i = tid; i < 64 * 128; i += 256) {   // V tile, transposed
            int r = i >> 7, d = i & 127;
            Vs[d * 64 + r] = Vg[(size_t)(s0 + r) * (HKV * D) + d];
        }
        wait_async0();
        __syncthreads();

        // S = Q K^T  (16x64 per wave: 4 N-tiles x 4 K-steps)
        v8f Sacc[4];
        for (int nt = 0; nt < 4; ++nt)
            Sacc[nt] = (v8f){0.f,0.f,0.f,0.f,0.f,0.f,0.f,0.f};
        for (int kk = 0; kk < 4; ++kk) {
            for (int nt = 0; nt < 4; ++nt) {
                Frag16 kf;
                const unsigned short* p = &Ks[(nt * 16 + lm) * 128 + kk * 32];
                kf.q[0] = *(const u32x4*)(p + 8 * hi);
                kf.q[1] = *(const u32x4*)(p + 16 + 8 * hi);
                Sacc[nt] = __builtin_amdgcn_wmma_f32_16x16x32_bf16(
                    false, qf[kk].v, false, kf.v, (short)0, Sacc[nt], false, false);
            }
        }

        // scale + causal mask
        for (int nt = 0; nt < 4; ++nt)
            for (int j = 0; j < 8; ++j) {
                int t = t0 + wid * 16 + j + 8 * hi;
                int s = s0 + nt * 16 + lm;
                float v = Sacc[nt][j] * sm;
                Sacc[nt][j] = (s <= t) ? v : -1e30f;
            }

        // online softmax (row stats per lane-half: rows j + 8*hi)
        float alpha[8];
        for (int j = 0; j < 8; ++j) {
            float mx = fmaxf(fmaxf(Sacc[0][j], Sacc[1][j]),
                             fmaxf(Sacc[2][j], Sacc[3][j]));
            for (int off = 1; off < 16; off <<= 1)
                mx = fmaxf(mx, __shfl_xor(mx, off, 32));
            float mnew = fmaxf(mrow[j], mx);
            alpha[j] = __expf(mrow[j] - mnew);
            mrow[j] = mnew;
        }
        for (int j = 0; j < 8; ++j) {
            float rs = 0.f;
            int rr = j + 8 * hi;
            for (int nt = 0; nt < 4; ++nt) {
                float p = __expf(Sacc[nt][j] - mrow[j]);
                rs += p;
                Ps[wid][rr * 64 + nt * 16 + lm] = f2bf(p);   // C-layout -> LDS
            }
            for (int off = 1; off < 16; off <<= 1)
                rs += __shfl_xor(rs, off, 32);
            lrow[j] = lrow[j] * alpha[j] + rs;
        }
        for (int n = 0; n < 8; ++n)
            for (int j = 0; j < 8; ++j)
                Oacc[n][j] *= alpha[j];

        // same-wave DS ops are in-order; wait for P writes before A-frag reads
        asm volatile("s_wait_dscnt 0" ::: "memory");

        // O += P V   (K=64 keys -> 2 WMMA k-steps, 8 d-tiles)
        for (int ks = 0; ks < 2; ++ks) {
            Frag16 pf;
            const unsigned short* pp = &Ps[wid][lm * 64 + ks * 32];
            pf.q[0] = *(const u32x4*)(pp + 8 * hi);
            pf.q[1] = *(const u32x4*)(pp + 16 + 8 * hi);
            for (int nt = 0; nt < 8; ++nt) {
                Frag16 vf;
                const unsigned short* vp = &Vs[(nt * 16 + lm) * 64 + ks * 32];
                vf.q[0] = *(const u32x4*)(vp + 8 * hi);
                vf.q[1] = *(const u32x4*)(vp + 16 + 8 * hi);
                Oacc[nt] = __builtin_amdgcn_wmma_f32_16x16x32_bf16(
                    false, pf.v, false, vf.v, (short)0, Oacc[nt], false, false);
            }
        }
    }

    // normalize rows and write bf16 context: Y[b*T+t][h*D + d]
    for (int j = 0; j < 8; ++j) {
        float inv = 1.f / lrow[j];
        int t = t0 + wid * 16 + j + 8 * hi;
        unsigned short* yrow = Y + ((size_t)(b * T + t)) * E + (size_t)h * D;
        for (int nt = 0; nt < 8; ++nt)
            yrow[nt * 16 + lm] = f2bf(Oacc[nt][j] * inv);
    }
}

// ---------------------------------------------------------------------------
extern "C" void kernel_launch(void* const* d_in, const int* in_sizes, int n_in,
                              void* d_out, int out_size, void* d_ws, size_t ws_size,
                              hipStream_t stream)
{
    const float* query = (const float*)d_in[0];
    const float* keyx  = (const float*)d_in[1];
    const float* value = (const float*)d_in[2];
    const float* fr    = (const float*)d_in[3];
    // d_in[4]: causal mask (int32) — causality is implemented analytically.
    const float* Wq = (const float*)d_in[5];  const float* bq = (const float*)d_in[6];
    const float* Wk = (const float*)d_in[7];  const float* bk = (const float*)d_in[8];
    const float* Wv = (const float*)d_in[9];  const float* bv = (const float*)d_in[10];
    const float* Wo = (const float*)d_in[11]; const float* bo = (const float*)d_in[12];

    const size_t M  = (size_t)Bsz * T;        // 4096 rows
    const size_t NK = (size_t)HKV * D;        // 1024

    char* w = (char*)d_ws;
    unsigned short* xbf = (unsigned short*)w;            w += M * E * 2;   // act staging
    unsigned short* wbf = (unsigned short*)w;            w += (size_t)E * E * 2; // wgt staging
    float*          Qp  = (float*)w;                     w += M * E * 4;
    float*          Kp  = (float*)w;                     w += M * NK * 4;
    float*          Vp  = (float*)w;                     w += M * NK * 4;
    unsigned short* Qbf = (unsigned short*)w;            w += M * E * 2;
    unsigned short* Kbf = (unsigned short*)w;            w += M * NK * 2;
    unsigned short* Vbf = (unsigned short*)w;            w += M * NK * 2;
    unsigned short* Ybf = xbf;    // reuse: activation staging is dead by then

    dim3 blk(256);
    const int nEE4 = (int)((size_t)E * E / 4);           // 4.19M quads
    const int nME4 = (int)(M * E / 4);
    const int nKW4 = (int)(NK * E / 4);

    // ---- Q projection
    cvt_f32_bf16_kernel<<<(nME4 + 255) / 256, blk, 0, stream>>>(query, xbf, nME4);
    cvt_f32_bf16_kernel<<<(nEE4 + 255) / 256, blk, 0, stream>>>(Wq, wbf, nEE4);
    gemm_nt_bf16<<<dim3(E / 128, (int)M / 128), blk, 0, stream>>>(
        xbf, wbf, bq, Qp, (int)M, E, E);
    // ---- K projection
    cvt_f32_bf16_kernel<<<(nME4 + 255) / 256, blk, 0, stream>>>(keyx, xbf, nME4);
    cvt_f32_bf16_kernel<<<(nKW4 + 255) / 256, blk, 0, stream>>>(Wk, wbf, nKW4);
    gemm_nt_bf16<<<dim3((int)NK / 128, (int)M / 128), blk, 0, stream>>>(
        xbf, wbf, bk, Kp, (int)M, (int)NK, E);
    // ---- V projection
    cvt_f32_bf16_kernel<<<(nME4 + 255) / 256, blk, 0, stream>>>(value, xbf, nME4);
    cvt_f32_bf16_kernel<<<(nKW4 + 255) / 256, blk, 0, stream>>>(Wv, wbf, nKW4);
    gemm_nt_bf16<<<dim3((int)NK / 128, (int)M / 128), blk, 0, stream>>>(
        xbf, wbf, bv, Vp, (int)M, (int)NK, E);

    // ---- RoPE (+ bf16) for Q, K; plain bf16 for V
    int totQ = Bsz * T * HQ  * (D / 2);
    int totK = Bsz * T * HKV * (D / 2);
    rope_to_bf16_kernel<<<(totQ + 255) / 256, blk, 0, stream>>>(Qp, Qbf, fr, HQ, totQ);
    rope_to_bf16_kernel<<<(totK + 255) / 256, blk, 0, stream>>>(Kp, Kbf, fr, HKV, totK);
    int nV4 = (int)(M * NK / 4);
    cvt_f32_bf16_kernel<<<(nV4 + 255) / 256, blk, 0, stream>>>(Vp, Vbf, nV4);

    // ---- attention
    flash_attn_kernel<<<dim3(T / 128, Bsz * HQ), blk, 0, stream>>>(Qbf, Kbf, Vbf, Ybf);

    // ---- output projection -> f32 d_out
    cvt_f32_bf16_kernel<<<(nEE4 + 255) / 256, blk, 0, stream>>>(Wo, wbf, nEE4);
    gemm_nt_bf16<<<dim3(E / 128, (int)M / 128), blk, 0, stream>>>(
        Ybf, wbf, bo, (float*)d_out, (int)M, E, E);
}
